// PagedAttention_9534827397285
// MI455X (gfx1250) — matrix operand order repaired and checked by reading, not compiled, not run on hardware
//
#include <hip/hip_runtime.h>
#include <cstdint>

typedef __attribute__((ext_vector_type(16))) _Float16 v16h;
typedef __attribute__((ext_vector_type(8)))  float    v8f;

// Problem constants (match reference)
static constexpr int kH  = 32;    // heads
static constexpr int kD  = 128;   // head dim
static constexpr int kLQ = 2048;  // query length
// kv advances in BN=32 tokens (2 pages of 16)

__global__ __launch_bounds__(256)
void paged_attn_wmma_kernel(const float* __restrict__ q,
                            const int8_t* __restrict__ Kp,
                            const int8_t* __restrict__ Vp,
                            const _Float16* __restrict__ Ksc,
                            const _Float16* __restrict__ Vsc,
                            const int* __restrict__ pages,
                            float* __restrict__ out)
{
    __shared__ _Float16 Kl[32 * 128];      // K tile, [token][d]      (8 KB)
    __shared__ _Float16 Vt[128 * 32];      // V tile, [d][token]      (8 KB)
    __shared__ _Float16 Pst[8 * 16 * 32];  // per-wave P staging      (8 KB)

    const int h    = blockIdx.y;
    const int bx   = blockIdx.x;
    const int tid  = threadIdx.x;
    const int wave = tid >> 5;
    const int lane = tid & 31;
    const int col  = lane & 15;       // N / row-piece selector
    const int hsel = lane >> 4;       // 0: lanes 0-15, 1: lanes 16-31

    const int qrow0 = bx * 128 + wave * 16;   // this wave's 16 query rows

    // ---------------- Q fragments (A-layout), fold in 1/sqrt(D) --------------
    // A 16x32 f16 layout: lane row M = lane&15; VGPR v holds
    // K = 16*(v>=4) + 8*(lane>=16) + 2*(v&3) (+1 in high half).
    const float qscale = 0.08838834764831845f; // 1/sqrt(128)
    const int   qrowA  = qrow0 + col;
    v16h Aq[4];
    #pragma unroll
    for (int s = 0; s < 4; ++s) {
        #pragma unroll
        for (int v = 0; v < 8; ++v) {
            const int k = ((v >= 4) ? 16 : 0) + hsel * 8 + 2 * (v & 3);
            const int d = 32 * s + k;
            const float* qp = q + ((h * kLQ + qrowA) * kD + d);
            Aq[s][2 * v]     = (_Float16)(qp[0] * qscale);
            Aq[s][2 * v + 1] = (_Float16)(qp[1] * qscale);
        }
    }

    // ---------------- accumulators -------------------------------------------
    v8f zero8 = {0.f, 0.f, 0.f, 0.f, 0.f, 0.f, 0.f, 0.f};
    v8f O[8];
    #pragma unroll
    for (int n = 0; n < 8; ++n) O[n] = zero8;
    float m_r[8], l_r[8];
    #pragma unroll
    for (int r = 0; r < 8; ++r) { m_r[r] = -__builtin_inff(); l_r[r] = 0.f; }

    const int kb_total = 4 * bx + 4;  // cover q up to bx*128+127 (causal)

    for (int kb = 0; kb < kb_total; ++kb) {
        const int kv0 = kb * 32;

        // ------------- cooperative stage: dequant int8 -> f16 into LDS -------
        {
            const int token = tid >> 3;           // 0..31
            const int dg    = (tid & 7) << 4;     // 0,16,..,112
            const int gt    = kv0 + token;
            const int pg    = pages[gt >> 4];
            const int tip   = gt & 15;
            const int base  = ((pg * 16 + tip) * kH + h) * kD + dg;
            const float ks  = (float)Ksc[pg * kH + h];
            const float vs  = (float)Vsc[pg * kH + h];
            const int4 kw = *(const int4*)(Kp + base);
            const int4 vw = *(const int4*)(Vp + base);
            const int kwa[4] = {kw.x, kw.y, kw.z, kw.w};
            const int vwa[4] = {vw.x, vw.y, vw.z, vw.w};
            #pragma unroll
            for (int j = 0; j < 4; ++j) {
                #pragma unroll
                for (int b = 0; b < 4; ++b) {
                    const int dd = dg + j * 4 + b;
                    const float fk = (float)((int8_t)(kwa[j] >> (8 * b)));
                    const float fv = (float)((int8_t)(vwa[j] >> (8 * b)));
                    Kl[token * 128 + dd] = (_Float16)(fk * ks);
                    Vt[dd * 32 + token]  = (_Float16)(fv * vs);
                }
            }
        }
        __syncthreads();

        // ------------- per-wave compute (causal skip is wave-uniform) --------
        if (kv0 <= qrow0 + 15) {
            // S = Q * K^T for 32 tokens -> two 16x16 C tiles
            v8f S0 = zero8, S1 = zero8;
            #pragma unroll
            for (int s = 0; s < 4; ++s) {
                v16h Bk0, Bk1;
                #pragma unroll
                for (int v = 0; v < 8; ++v) {
                    // B 32x16 layout: N = lane&15 (token), K(d) = 16*(v>=4) + 8*hsel + 2*(v&3)
                    const int d = 32 * s + ((v >= 4) ? 16 : 0) + hsel * 8 + 2 * (v & 3);
                    Bk0[2 * v]     = Kl[col * 128 + d];
                    Bk0[2 * v + 1] = Kl[col * 128 + d + 1];
                    Bk1[2 * v]     = Kl[(16 + col) * 128 + d];
                    Bk1[2 * v + 1] = Kl[(16 + col) * 128 + d + 1];
                }
                S0 = __builtin_amdgcn_wmma_f32_16x16x32_f16(false, Aq[s], false, Bk0,
                                                            (short)0, S0, false, false);
                S1 = __builtin_amdgcn_wmma_f32_16x16x32_f16(false, Aq[s], false, Bk1,
                                                            (short)0, S1, false, false);
            }

            // online softmax on C layout: row M = r + 8*hsel, N = col
            const int t0g = kv0 + col, t1g = t0g + 16;
            float p0[8], p1[8], alpha[8];
            #pragma unroll
            for (int r = 0; r < 8; ++r) {
                const int qi = qrow0 + r + hsel * 8;
                float s0 = (t0g <= qi) ? S0[r] : -__builtin_inff();
                float s1 = (t1g <= qi) ? S1[r] : -__builtin_inff();
                float x = fmaxf(s0, s1);
                x = fmaxf(x, __shfl_xor(x, 1, 32));
                x = fmaxf(x, __shfl_xor(x, 2, 32));
                x = fmaxf(x, __shfl_xor(x, 4, 32));
                x = fmaxf(x, __shfl_xor(x, 8, 32));
                const float nm = fmaxf(m_r[r], x);
                alpha[r] = __expf(m_r[r] - nm);
                p0[r]    = __expf(s0 - nm);
                p1[r]    = __expf(s1 - nm);
                float rs = p0[r] + p1[r];
                rs += __shfl_xor(rs, 1, 32);
                rs += __shfl_xor(rs, 2, 32);
                rs += __shfl_xor(rs, 4, 32);
                rs += __shfl_xor(rs, 8, 32);
                l_r[r] = l_r[r] * alpha[r] + rs;
                m_r[r] = nm;
            }
            #pragma unroll
            for (int n = 0; n < 8; ++n)
                #pragma unroll
                for (int r = 0; r < 8; ++r) O[n][r] *= alpha[r];

            // P: C layout -> A layout via per-wave LDS staging
            _Float16* pw = Pst + wave * 512;
            #pragma unroll
            for (int r = 0; r < 8; ++r) {
                const int M = r + hsel * 8;
                pw[M * 32 + col]      = (_Float16)p0[r];
                pw[M * 32 + 16 + col] = (_Float16)p1[r];
            }
            asm volatile("s_wait_dscnt 0" ::: "memory");
            v16h Pa;
            #pragma unroll
            for (int v = 0; v < 8; ++v) {
                const int k = ((v >= 4) ? 16 : 0) + hsel * 8 + 2 * (v & 3);
                Pa[2 * v]     = pw[col * 32 + k];
                Pa[2 * v + 1] = pw[col * 32 + k + 1];
            }

            // O += P * V   (8 d-tiles of 16 columns)
            #pragma unroll
            for (int n = 0; n < 8; ++n) {
                v16h Bv;
                const int d = 16 * n + col;   // B: N = d column
                #pragma unroll
                for (int v = 0; v < 8; ++v) {
                    const int t = ((v >= 4) ? 16 : 0) + hsel * 8 + 2 * (v & 3);   // B: K = token (table order)
                    Bv[2 * v]     = Vt[d * 32 + t];
                    Bv[2 * v + 1] = Vt[d * 32 + t + 1];
                }
                O[n] = __builtin_amdgcn_wmma_f32_16x16x32_f16(false, Pa, false, Bv,
                                                              (short)0, O[n], false, false);
            }
        }
        __syncthreads();
    }

    // ---------------- epilogue: normalize and store --------------------------
    float inv[8];
    #pragma unroll
    for (int r = 0; r < 8; ++r) inv[r] = 1.0f / l_r[r];
    #pragma unroll
    for (int n = 0; n < 8; ++n) {
        const int d = 16 * n + col;
        #pragma unroll
        for (int r = 0; r < 8; ++r) {
            const int row = qrow0 + r + hsel * 8;
            out[(h * kLQ + row) * kD + d] = O[n][r] * inv[r];
        }
    }
}

extern "C" void kernel_launch(void* const* d_in, const int* in_sizes, int n_in,
                              void* d_out, int out_size, void* d_ws, size_t ws_size,
                              hipStream_t stream) {
    (void)in_sizes; (void)n_in; (void)out_size; (void)d_ws; (void)ws_size;
    const float*    qp  = (const float*)d_in[0];
    const int8_t*   Kp  = (const int8_t*)d_in[1];
    const int8_t*   Vp  = (const int8_t*)d_in[2];
    const _Float16* Ks  = (const _Float16*)d_in[3];
    const _Float16* Vs  = (const _Float16*)d_in[4];
    const int*      pg  = (const int*)d_in[5];
    float*          op  = (float*)d_out;

    dim3 grid(kLQ / 128, kH);   // 16 query-blocks x 32 heads
    paged_attn_wmma_kernel<<<grid, 256, 0, stream>>>(qp, Kp, Vp, Ks, Vs, pg, op);
}
